// MoELayer_63848983823107
// MI455X (gfx1250) — matrix operand order repaired and checked by reading, not compiled, not run on hardware
//
#include <hip/hip_runtime.h>

// ---------------- problem constants (from reference) ----------------
#define N_TOK   32768
#define D_MODEL 256
#define D_FF    1024
#define N_EXP   8

// ---------------- tiling ----------------
#define TILE_N   64      // tokens per block
#define FB       64      // DFF block per iteration
#define NITER    (D_FF / FB)
#define NTHREADS 256     // 8 waves of 32

typedef __attribute__((ext_vector_type(16))) __bf16 v16bf;
typedef __attribute__((ext_vector_type(8)))  float  v8f;
typedef unsigned int u32x4 __attribute__((ext_vector_type(4)));
typedef int          i32x4 __attribute__((ext_vector_type(4)));
typedef int          i32x8 __attribute__((ext_vector_type(8)));

union FragBF { v16bf v; unsigned u[8]; };

// ---------------- LDS layout of the expert kernel (byte offsets) ----------------
#define LDS_XS    0                        // [64][256] bf16 x tile        (32 KB)
#define LDS_W1    32768                    // [FB][256] bf16 w1 slice x2   (64 KB)
#define LDS_W2    98304                    // [256][FB] bf16 w2 slice x2   (64 KB)
#define LDS_HS    163840                   // [64][FB] bf16 hidden          (8 KB)
#define LDS_TID   172032                   // [64] int token ids
#define LDS_TW    172288                   // [64] float gate weights
#define LDS_TOTAL 172544
#define WBUF_BYTES 32768                   // one weight tile buffer

// f32 -> bf16 (round to nearest even), pure bit ops
__device__ __forceinline__ unsigned short f32_to_bf16_raw(float f) {
    unsigned u = __float_as_uint(f);
    unsigned r = (u + 0x7FFFu + ((u >> 16) & 1u)) >> 16;
    return (unsigned short)r;
}
__device__ __forceinline__ unsigned pack_bf16x2(float lo, float hi) {
    return (unsigned)f32_to_bf16_raw(lo) | ((unsigned)f32_to_bf16_raw(hi) << 16);
}
// generic -> 32-bit LDS byte offset (LDS generic addr low 32 bits == LDS_ADDR)
__device__ __forceinline__ unsigned lds_off(const void* p) {
    return (unsigned)(unsigned long long)p;
}

// A-matrix fragment, 16x32 bf16, contiguous k order (same as B): lane L: M=L%16;
// VGPR j: K = 2j + (L>=16 ? 16 : 0). LDS layout [M][K] row-major.
__device__ __forceinline__ v16bf load_frag_a(const unsigned short* lds, int k0, int ld) {
    int lane   = threadIdx.x & 31;
    int m      = lane & 15;
    int kpair4 = (lane >> 4) * 8;
    FragBF r;
    const unsigned* base = (const unsigned*)(lds + (size_t)m * ld + k0);
#pragma unroll
    for (int j = 0; j < 8; ++j) {
        int pair = j + kpair4;
        r.u[j] = base[pair];
    }
    return r.v;
}
// B-matrix fragment, 32x16 bf16: lane L: N=L%16; VGPR j: K = 2j + (L>=16 ? 16 : 0).
// LDS layout transposed [N][K] (K contiguous) -> each VGPR one aligned b32 load.
__device__ __forceinline__ v16bf load_frag_b(const unsigned short* lds, int n0, int k0, int ldk) {
    int lane = threadIdx.x & 31;
    int n    = lane & 15;
    int kh   = (lane >> 4) << 4;
    FragBF r;
    const unsigned* base = (const unsigned*)(lds + (size_t)(n0 + n) * ldk + k0 + kh);
#pragma unroll
    for (int j = 0; j < 8; ++j) r.u[j] = base[j];
    return r.v;
}

// ---- Tensor Data Mover: 2D tile (bf16 elements) global -> LDS (ISA ch.8 D# layout) ----
__device__ __forceinline__ void tdm_load_2d_bf16(unsigned lds_addr, const void* gaddr,
                                                 unsigned tensor_d0, unsigned tensor_d1,
                                                 unsigned tile_d0, unsigned tile_d1,
                                                 unsigned long long stride0_elems) {
    unsigned long long ga = (unsigned long long)gaddr;
    u32x4 g0;
    g0[0] = 0x1u;                                   // count=1, user mode, no gather
    g0[1] = lds_addr;                               // LDS byte address
    g0[2] = (unsigned)(ga & 0xFFFFFFFFu);           // global addr [31:0]
    g0[3] = (unsigned)((ga >> 32) & 0x01FFFFFFu)    // global addr [56:32]
            | (2u << 30);                           // type = 2 ("image")
    i32x8 g1;
    g1[0] = 0x00010000;                             // workgroup_mask=0, data_size=1 (2B)
    g1[1] = (int)((tensor_d0 & 0xFFFFu) << 16);     // [63:48] tensor_dim0 lo16
    g1[2] = (int)((tensor_d0 >> 16) | ((tensor_d1 & 0xFFFFu) << 16));
    g1[3] = (int)((tensor_d1 >> 16) | (tile_d0 << 16));   // [127:112] tile_dim0
    g1[4] = (int)(tile_d1 & 0xFFFFu);               // tile_dim1; tile_dim2=0
    g1[5] = (int)(stride0_elems & 0xFFFFFFFFu);     // tensor_dim0_stride lo32
    g1[6] = (int)((stride0_elems >> 32) & 0xFFFFu); // stride hi16; dim1_stride=0
    g1[7] = 0;
    i32x4 z = {0, 0, 0, 0};
#if defined(__clang_major__) && (__clang_major__ >= 23)
    i32x8 z8 = {0, 0, 0, 0, 0, 0, 0, 0};
    __builtin_amdgcn_tensor_load_to_lds(g0, g1, z, z, z8, 0);
#else
    __builtin_amdgcn_tensor_load_to_lds(g0, g1, z, z, 0);
#endif
}

// ---------------- kernel 0: zero output + expert counters ----------------
__global__ void moe_zero_kernel(float* __restrict__ out, int* __restrict__ counts, long total4) {
    long i = (long)blockIdx.x * blockDim.x + threadIdx.x;
    if (i < total4) {
        float4 z; z.x = 0.f; z.y = 0.f; z.z = 0.f; z.w = 0.f;
        ((float4*)out)[i] = z;
    }
    if (i < N_EXP) counts[i] = 0;
}

// ---------------- prep: x -> bf16 ----------------
__global__ void moe_cvt_x(const float* __restrict__ x, unsigned* __restrict__ xb2, long n4) {
    long i = (long)blockIdx.x * blockDim.x + threadIdx.x;
    if (i >= n4) return;
    float4 v = ((const float4*)x)[i];
    xb2[i * 2 + 0] = pack_bf16x2(v.x, v.y);
    xb2[i * 2 + 1] = pack_bf16x2(v.z, v.w);
}

// ---------------- prep: weight transpose + bf16: dst[e][c][r] = src[e][r][c] ----------------
__global__ void moe_cvt_wT(const float* __restrict__ src, unsigned short* __restrict__ dst,
                           int R, int C) {
    long j = (long)blockIdx.x * blockDim.x + threadIdx.x;
    long rc = (long)R * C;
    if (j >= (long)N_EXP * rc) return;
    int e = (int)(j / rc);
    long rem = j - (long)e * rc;
    int c = (int)(rem / R);
    int r = (int)(rem - (long)c * R);
    dst[j] = f32_to_bf16_raw(src[(long)e * rc + (long)r * C + c]);
}

// ---------------- kernel 1: gating (softmax + top-2) + routing scatter ----------------
__global__ void moe_gate_kernel(const float* __restrict__ x,
                                const float* __restrict__ gate_w,
                                const float* __restrict__ gate_b,
                                int* __restrict__ counts,
                                int* __restrict__ tok_id,
                                float* __restrict__ tok_w) {
    __shared__ float gws[D_MODEL * N_EXP];
    for (int i = threadIdx.x; i < D_MODEL * N_EXP; i += blockDim.x) gws[i] = gate_w[i];
    __syncthreads();

    int n = blockIdx.x * blockDim.x + threadIdx.x;
    float l[N_EXP];
#pragma unroll
    for (int e = 0; e < N_EXP; ++e) l[e] = gate_b[e];

    const float4* xr = (const float4*)(x + (size_t)n * D_MODEL);
    for (int d4 = 0; d4 < D_MODEL / 4; ++d4) {
        float4 xv = xr[d4];
        const float* g0 = &gws[(d4 * 4) * N_EXP];
#pragma unroll
        for (int e = 0; e < N_EXP; ++e)
            l[e] += xv.x * g0[e] + xv.y * g0[N_EXP + e] +
                    xv.z * g0[2 * N_EXP + e] + xv.w * g0[3 * N_EXP + e];
    }
    float m = l[0];
#pragma unroll
    for (int e = 1; e < N_EXP; ++e) m = fmaxf(m, l[e]);
    float p[N_EXP], s = 0.f;
#pragma unroll
    for (int e = 0; e < N_EXP; ++e) { p[e] = __expf(l[e] - m); s += p[e]; }
    float inv = 1.f / s;
    int i0 = 0;
#pragma unroll
    for (int e = 1; e < N_EXP; ++e) if (p[e] > p[i0]) i0 = e;
    int i1 = (i0 == 0) ? 1 : 0;
#pragma unroll
    for (int e = 0; e < N_EXP; ++e) if (e != i0 && p[e] > p[i1]) i1 = e;

    int s0 = atomicAdd(&counts[i0], 1);
    tok_id[i0 * N_TOK + s0] = n;
    tok_w [i0 * N_TOK + s0] = p[i0] * inv;
    int s1 = atomicAdd(&counts[i1], 1);
    tok_id[i1 * N_TOK + s1] = n;
    tok_w [i1 * N_TOK + s1] = p[i1] * inv;
}

// ---------------- kernel 2: routed expert FFN, bf16 WMMA + TDM + async gather ----------------
__global__ void __launch_bounds__(NTHREADS)
moe_expert_kernel(const unsigned short* __restrict__ xb,    // [N][256] bf16
                  const unsigned short* __restrict__ w1t,   // [E][DFF][256] bf16 (transposed)
                  const unsigned short* __restrict__ w2t,   // [E][256][DFF] bf16 (transposed)
                  const float* __restrict__ bias1, const float* __restrict__ bias2,
                  const int* __restrict__ counts,
                  const int* __restrict__ tok_id, const float* __restrict__ tok_w,
                  float* __restrict__ out) {
    extern __shared__ char smem[];
    unsigned short* xs = (unsigned short*)(smem + LDS_XS);
    unsigned short* hs = (unsigned short*)(smem + LDS_HS);
    int*   tids = (int*)(smem + LDS_TID);
    float* tws  = (float*)(smem + LDS_TW);
    const unsigned smem_lds = lds_off(smem);   // LDS byte base for TDM descriptors

    const int e     = blockIdx.y;
    const int cnt   = counts[e];
    const int tile0 = blockIdx.x * TILE_N;
    if (tile0 >= cnt) return;

    const int tid  = threadIdx.x;
    const int lane = tid & 31;
    const int wave = tid >> 5;

    const unsigned short* w1e = w1t + (size_t)e * D_FF * D_MODEL;
    const unsigned short* w2e = w2t + (size_t)e * D_MODEL * D_FF;

    // prefetch weight tiles for fb=0 via Tensor Data Mover (wave 0 issues)
    if (wave == 0) {
        tdm_load_2d_bf16(smem_lds + LDS_W1, w1e, D_MODEL, D_FF, D_MODEL, FB, D_MODEL);
        tdm_load_2d_bf16(smem_lds + LDS_W2, w2e, D_FF, D_MODEL, FB, D_MODEL, D_FF);
    }

    // gather token ids / gate weights
    if (tid < TILE_N) {
        int slot = tile0 + tid;
        int cs   = slot < cnt ? slot : cnt - 1;      // pad with last token, gate weight 0
        tids[tid] = tok_id[e * N_TOK + cs];
        tws[tid]  = (slot < cnt) ? tok_w[e * N_TOK + cs] : 0.f;
    }
    __syncthreads();

    // gather x rows (bf16, 512 B/row) straight into LDS: async copy, per-lane addresses
    {
        const unsigned laneB = (unsigned)lane * 16u;
#pragma unroll
        for (int r = 0; r < 8; ++r) {
            int row = wave * 8 + r;
            unsigned voff = (unsigned)tids[row] * 512u + laneB;
            unsigned ldsa = smem_lds + LDS_XS + (unsigned)row * 512u + laneB;
            asm volatile("global_load_async_to_lds_b128 %0, %1, %2"
                         :: "v"(ldsa), "v"(voff), "s"(xb) : "memory");
        }
        asm volatile("s_wait_asynccnt 0x0" ::: "memory");
    }

    const int g    = wave >> 1;          // token group 0..3 (16 tokens)
    const int half = wave & 1;           // DFF-half (GEMM1) / D-out-half (GEMM2)
    const int nIdx = lane & 15;
    const int mAdd = (lane >> 4) * 8;

    v8f acc[8];
#pragma unroll
    for (int t = 0; t < 8; ++t)
#pragma unroll
        for (int r = 0; r < 8; ++r) acc[t][r] = 0.f;

    for (int it = 0; it < NITER; ++it) {
        const int fb  = it * FB;
        const int buf = it & 1;
        const unsigned short* w1c = (const unsigned short*)(smem + LDS_W1 + (size_t)buf * WBUF_BYTES);
        const unsigned short* w2c = (const unsigned short*)(smem + LDS_W2 + (size_t)buf * WBUF_BYTES);
        __syncthreads();    // previous iteration's compute done (frees buf^1 and hs)

        if (wave == 0) {
            if (it + 1 < NITER) {   // prefetch next tiles into the other buffer
                const int fn = fb + FB;
                const unsigned nb = (unsigned)(buf ^ 1) * WBUF_BYTES;
                tdm_load_2d_bf16(smem_lds + LDS_W1 + nb, w1e + (size_t)fn * D_MODEL,
                                 D_MODEL, D_FF, D_MODEL, FB, D_MODEL);
                tdm_load_2d_bf16(smem_lds + LDS_W2 + nb, w2e + fn,
                                 D_FF, D_MODEL, FB, D_MODEL, D_FF);
                __builtin_amdgcn_s_wait_tensorcnt(2);   // current buf's pair complete
            } else {
                __builtin_amdgcn_s_wait_tensorcnt(0);
            }
        }
        __syncthreads();    // publish current weight tiles to all waves

        // GEMM1: h[g*16..+15, f0..f0+31] = x_tile @ w1 ; +bias ; ReLU -> hs (bf16)
        {
            const int f0 = half * 32;
            v8f c0, c1;
#pragma unroll
            for (int r = 0; r < 8; ++r) { c0[r] = 0.f; c1[r] = 0.f; }
#pragma unroll
            for (int k = 0; k < D_MODEL; k += 32) {
                v16bf a  = load_frag_a(xs + (g * 16) * D_MODEL, k, D_MODEL);
                v16bf b0 = load_frag_b(w1c, f0,      k, D_MODEL);
                v16bf bX = load_frag_b(w1c, f0 + 16, k, D_MODEL);
                c0 = __builtin_amdgcn_wmma_f32_16x16x32_bf16(false, a, false, b0, (short)0, c0, false, false);
                c1 = __builtin_amdgcn_wmma_f32_16x16x32_bf16(false, a, false, bX, (short)0, c1, false, false);
            }
            float bi0 = bias1[e * D_FF + fb + f0 + nIdx];
            float bi1 = bias1[e * D_FF + fb + f0 + 16 + nIdx];
#pragma unroll
            for (int r = 0; r < 8; ++r) {
                int m = r + mAdd;
                float h0 = fmaxf(c0[r] + bi0, 0.f);
                float h1 = fmaxf(c1[r] + bi1, 0.f);
                hs[(g * 16 + m) * FB + f0      + nIdx] = f32_to_bf16_raw(h0);
                hs[(g * 16 + m) * FB + f0 + 16 + nIdx] = f32_to_bf16_raw(h1);
            }
        }
        __syncthreads();

        // GEMM2: acc[g*16..+15, d0..d0+127] += h @ w2
        {
            const int d0 = half * 128;
#pragma unroll
            for (int k = 0; k < FB; k += 32) {
                v16bf a = load_frag_a(hs + (g * 16) * FB, k, FB);
#pragma unroll
                for (int t = 0; t < 8; ++t) {
                    v16bf b = load_frag_b(w2c, d0 + t * 16, k, FB);
                    acc[t] = __builtin_amdgcn_wmma_f32_16x16x32_bf16(false, a, false, b, (short)0, acc[t], false, false);
                }
            }
        }
    }

    // epilogue: out[token, d] += gate_w * (acc + b2[d])
    float gw[8]; int tk[8];
#pragma unroll
    for (int r = 0; r < 8; ++r) {
        int tl = g * 16 + r + mAdd;
        gw[r] = tws[tl];
        tk[r] = tids[tl];
    }
    const int d0 = half * 128;
#pragma unroll
    for (int t = 0; t < 8; ++t) {
        int d = d0 + t * 16 + nIdx;
        float b2v = bias2[e * D_MODEL + d];
#pragma unroll
        for (int r = 0; r < 8; ++r) {
            atomicAdd(out + (size_t)tk[r] * D_MODEL + d, gw[r] * (acc[t][r] + b2v));
        }
    }
}

// ---------------- launcher ----------------
extern "C" void kernel_launch(void* const* d_in, const int* in_sizes, int n_in,
                              void* d_out, int out_size, void* d_ws, size_t ws_size,
                              hipStream_t stream) {
    const float* x      = (const float*)d_in[0];
    const float* gate_w = (const float*)d_in[1];
    const float* gate_b = (const float*)d_in[2];
    const float* w1     = (const float*)d_in[3];
    const float* b1     = (const float*)d_in[4];
    const float* w2     = (const float*)d_in[5];
    const float* b2     = (const float*)d_in[6];
    float* out = (float*)d_out;

    // workspace: counts | tok_id | tok_w | xb(bf16) | w1t(bf16) | w2t(bf16)
    char* ws = (char*)d_ws;
    int*   counts = (int*)ws;
    int*   tok_id = (int*)(ws + 256);
    float* tok_w  = (float*)(ws + 256 + (size_t)N_EXP * N_TOK * 4);
    unsigned short* xb  = (unsigned short*)(ws + 256 + (size_t)N_EXP * N_TOK * 8);
    unsigned short* w1t = xb  + (size_t)N_TOK * D_MODEL;
    unsigned short* w2t = w1t + (size_t)N_EXP * D_FF * D_MODEL;

    // 0) zero output + counters
    long total4 = (long)N_TOK * D_MODEL / 4;
    moe_zero_kernel<<<(int)((total4 + 255) / 256), 256, 0, stream>>>(out, counts, total4);

    // prep: bf16 conversions / transposes
    moe_cvt_x<<<(int)((total4 + 255) / 256), 256, 0, stream>>>(x, (unsigned*)xb, total4);
    long wtot = (long)N_EXP * D_MODEL * D_FF;
    moe_cvt_wT<<<(int)((wtot + 255) / 256), 256, 0, stream>>>(w1, w1t, D_MODEL, D_FF); // [e][f][k]
    moe_cvt_wT<<<(int)((wtot + 255) / 256), 256, 0, stream>>>(w2, w2t, D_FF, D_MODEL); // [e][d][f]

    // 1) gating + routing
    moe_gate_kernel<<<N_TOK / 256, 256, 0, stream>>>(x, gate_w, gate_b, counts, tok_id, tok_w);

    // 2) routed expert FFN
    (void)hipFuncSetAttribute((const void*)moe_expert_kernel,
                              hipFuncAttributeMaxDynamicSharedMemorySize, LDS_TOTAL);
    dim3 grid(N_TOK / TILE_N, N_EXP);
    moe_expert_kernel<<<grid, NTHREADS, LDS_TOTAL, stream>>>(
        xb, w1t, w2t, b1, b2, counts, tok_id, tok_w, out);
}
